// QuantumTransformerBlock_59004260712622
// MI455X (gfx1250) — matrix operand order repaired and checked by reading, not compiled, not run on hardware
//
#include <hip/hip_runtime.h>
#include <math.h>

#define BATCH   2
#define SEQ     2048
#define DIM     1024
#define NHEADS  16
#define HD      64
#define FFDIM   4096
#define MROWS   (BATCH * SEQ)   // 4096

typedef __attribute__((ext_vector_type(16))) _Float16 v16h;
typedef __attribute__((ext_vector_type(8)))  _Float16 v8h;
typedef __attribute__((ext_vector_type(8)))  float    v8f;

// Concatenate two 8-half LDS/global chunks into a 16-half WMMA fragment.
__device__ __forceinline__ v16h frag_cat(v8h lo, v8h hi) {
    return __builtin_shufflevector(lo, hi, 0, 1, 2, 3, 4, 5, 6, 7,
                                           8, 9, 10, 11, 12, 13, 14, 15);
}
__device__ __forceinline__ v8h ld8(const _Float16* p) {
    return *(const v8h*)p;
}

// gfx1250 async global->LDS 16-byte copy (ASYNCcnt tracked).
__device__ __forceinline__ void async_b128(const _Float16* gsrc, _Float16* ldst) {
    unsigned l = (unsigned)(uintptr_t)ldst;
    asm volatile("global_load_async_to_lds_b128 %0, %1, off"
                 :: "v"(l), "v"(gsrc) : "memory");
}
__device__ __forceinline__ void wait_async0() {
    asm volatile("s_wait_asynccnt 0x0" ::: "memory");
}

// ---------------------------------------------------------------------------
// LayerNorm over rows of [M, DIM], output f16 for WMMA consumption.
__global__ __launch_bounds__(256)
void layernorm_f16_kernel(const float* __restrict__ x,
                          const float* __restrict__ g,
                          const float* __restrict__ b,
                          _Float16* __restrict__ out) {
    __shared__ float s_sum[256];
    __shared__ float s_sq[256];
    const int tid = threadIdx.x;
    const size_t row = blockIdx.x;
    const float* xr = x + row * DIM;
    float lsum = 0.f, lsq = 0.f;
    #pragma unroll
    for (int i = tid; i < DIM; i += 256) {
        float v = xr[i];
        lsum += v; lsq += v * v;
    }
    s_sum[tid] = lsum; s_sq[tid] = lsq;
    __syncthreads();
    for (int s = 128; s > 0; s >>= 1) {
        if (tid < s) { s_sum[tid] += s_sum[tid + s]; s_sq[tid] += s_sq[tid + s]; }
        __syncthreads();
    }
    const float mu  = s_sum[0] * (1.0f / DIM);
    const float var = s_sq[0] * (1.0f / DIM) - mu * mu;
    const float rs  = rsqrtf(var + 1e-5f);
    _Float16* orow = out + row * DIM;
    #pragma unroll
    for (int i = tid; i < DIM; i += 256)
        orow[i] = (_Float16)((xr[i] - mu) * rs * g[i] + b[i]);
}

// ---------------------------------------------------------------------------
// WMMA GEMM:  C[M,N] = A_f16[M,K] * W_f32[K,N] + bias
// A tile async-copied to LDS; W tile converted f32->f16 and staged TRANSPOSED
// so both fragment types are contiguous 2x16B LDS loads per lane.
// Block tile 128x64, 8 waves (4x2), each wave 32x32 (2x2 wmma tiles).
// EPI: 0 = identity, 1 = exact GELU.
template <int EPI>
__global__ __launch_bounds__(256)
void gemm_f16_kernel(const _Float16* __restrict__ A,
                     const float* __restrict__ W,
                     const float* __restrict__ bias,
                     float* __restrict__ outF,
                     _Float16* __restrict__ outH,
                     int K, int N) {
    __shared__ _Float16 As[128][48];   // [m][k], stride 96B (16B aligned)
    __shared__ _Float16 Bt[64][48];    // [n][k] transposed, stride 96B

    const int tid  = threadIdx.x;
    const int lane = tid & 31;
    const int wid  = tid >> 5;
    const int wm   = wid >> 1;       // 0..3
    const int wn   = wid & 1;        // 0..1
    const int lg   = lane >> 4;
    const int l16  = lane & 15;
    const int bm   = blockIdx.y * 128;
    const int bn   = blockIdx.x * 64;

    v8f acc[2][2] = {};

    for (int k0 = 0; k0 < K; k0 += 32) {
        // ---- async-stage A tile: 128x32 halves = 512 x 16B chunks ----
        #pragma unroll
        for (int j = 0; j < 2; ++j) {
            int lin = tid + j * 256;
            int r = lin >> 2;
            int c = (lin & 3) * 8;
            async_b128(A + ((size_t)(bm + r) * K + k0 + c), &As[r][c]);
        }
        // ---- stage W tile transposed: 32x64 f32 -> f16, [n][k] ----
        #pragma unroll
        for (int j = 0; j < 2; ++j) {
            int lin = tid + j * 256;
            int r = lin >> 4;            // k row 0..31
            int c = (lin & 15) * 4;      // n col 0..60
            float4 t = *(const float4*)(W + ((size_t)(k0 + r) * N + bn + c));
            Bt[c + 0][r] = (_Float16)t.x;
            Bt[c + 1][r] = (_Float16)t.y;
            Bt[c + 2][r] = (_Float16)t.z;
            Bt[c + 3][r] = (_Float16)t.w;
        }
        if (k0 + 32 < K) {  // gfx1250 global_prefetch_b8 of next W tile
            __builtin_prefetch(W + ((size_t)(k0 + 32 + (tid >> 3)) * N + bn), 0, 1);
        }
        wait_async0();
        __syncthreads();

        v16h a[2], bfr[2];
        #pragma unroll
        for (int sm = 0; sm < 2; ++sm) {
            const _Float16* ar = &As[wm * 32 + sm * 16 + l16][0];
            a[sm] = frag_cat(ld8(ar + lg * 8), ld8(ar + 16 + lg * 8));
        }
        #pragma unroll
        for (int sn = 0; sn < 2; ++sn) {
            const _Float16* br = &Bt[wn * 32 + sn * 16 + l16][0];
            bfr[sn] = frag_cat(ld8(br + lg * 8), ld8(br + 16 + lg * 8));
        }
        #pragma unroll
        for (int sm = 0; sm < 2; ++sm)
            #pragma unroll
            for (int sn = 0; sn < 2; ++sn)
                acc[sm][sn] = __builtin_amdgcn_wmma_f32_16x16x32_f16(
                    false, a[sm], false, bfr[sn], (short)0, acc[sm][sn], false, false);
        __syncthreads();
    }

    // ---- epilogue ----
    #pragma unroll
    for (int sm = 0; sm < 2; ++sm) {
        #pragma unroll
        for (int sn = 0; sn < 2; ++sn) {
            const int col = bn + wn * 32 + sn * 16 + l16;
            const float bv = bias[col];
            #pragma unroll
            for (int r = 0; r < 8; ++r) {
                int row = bm + wm * 32 + sm * 16 + (lg ? 8 + r : r);
                float v = acc[sm][sn][r] + bv;
                if (EPI == 1) v = 0.5f * v * (1.0f + erff(v * 0.70710678118654752f));
                size_t idx = (size_t)row * N + col;
                if (outF) outF[idx] = v;
                if (outH) outH[idx] = (_Float16)v;
            }
        }
    }
}

// ---------------------------------------------------------------------------
// Flash attention with cosine score modulation.
// grid = (SEQ/128, NHEADS, BATCH); 8 waves/block, each wave owns 16 query rows.
// K tile async-copied [key][dim]; V tile staged transposed [dim][key].
__global__ __launch_bounds__(256)
void flash_attn_kernel(const _Float16* __restrict__ Q,
                       const _Float16* __restrict__ Km,
                       const _Float16* __restrict__ V,
                       const float* __restrict__ qphase,
                       _Float16* __restrict__ O) {
    __shared__ _Float16 Ks[32][80];     // [key][dim], stride 160B
    __shared__ _Float16 Vt[64][48];     // [dim][key], stride 96B
    __shared__ _Float16 Ps[8][16][40];  // per-wave P, stride 80B

    const int tid  = threadIdx.x;
    const int lane = tid & 31;
    const int wid  = tid >> 5;
    const int lg   = lane >> 4;
    const int l16  = lane & 15;
    const int bb   = blockIdx.z;
    const int hh   = blockIdx.y;
    const int q0   = blockIdx.x * 128 + wid * 16;
    const float phase = qphase[hh];
    const float scale = 0.125f;  // 1/sqrt(64)

    // Load Q fragments once (two K=32 chunks of head-dim 64), vectorized.
    v16h aq[2];
    const _Float16* qp = Q + ((size_t)bb * SEQ + q0 + l16) * DIM + hh * HD;
    #pragma unroll
    for (int c = 0; c < 2; ++c)
        aq[c] = frag_cat(ld8(qp + c * 32 + lg * 8),
                         ld8(qp + c * 32 + 16 + lg * 8));

    float mrow[8], lrow[8];
    #pragma unroll
    for (int r = 0; r < 8; ++r) { mrow[r] = -1e30f; lrow[r] = 0.f; }
    v8f o[4] = {};

    for (int kt = 0; kt < SEQ / 32; ++kt) {
        __syncthreads();
        {   // stage K/V tile: 32 keys x 64 dims
            int r = tid >> 3;            // key 0..31
            int c = (tid & 7) * 8;       // dim 0..56
            size_t base = ((size_t)bb * SEQ + kt * 32 + r) * DIM + hh * HD + c;
            async_b128(Km + base, &Ks[r][c]);          // K stays [key][dim]
            float4 tv = *(const float4*)(V + base);    // V transposed to [dim][key]
            const _Float16* hv = (const _Float16*)&tv;
            #pragma unroll
            for (int i = 0; i < 8; ++i) Vt[c + i][r] = hv[i];
        }
        wait_async0();
        __syncthreads();

        // scores: 16 rows x 32 keys (K-dim of wmma = head-dim, contiguous in Ks)
        v8f s[2] = {};
        #pragma unroll
        for (int sn = 0; sn < 2; ++sn) {
            const _Float16* kr = &Ks[sn * 16 + l16][0];
            #pragma unroll
            for (int c = 0; c < 2; ++c) {
                v16h bk = frag_cat(ld8(kr + c * 32 + lg * 8),
                                   ld8(kr + c * 32 + 16 + lg * 8));
                s[sn] = __builtin_amdgcn_wmma_f32_16x16x32_f16(
                    false, aq[c], false, bk, (short)0, s[sn], false, false);
            }
        }

        // modulation + online softmax (per-row stats via intra-half shuffles)
        float alpha[8];
        #pragma unroll
        for (int r = 0; r < 8; ++r) {
            float v0 = s[0][r] * scale;
            float v1 = s[1][r] * scale;
            v0 = v0 * (1.f + 0.1f * __cosf(phase * v0));
            v1 = v1 * (1.f + 0.1f * __cosf(phase * v1));
            float tm = fmaxf(v0, v1);
            #pragma unroll
            for (int off = 1; off < 16; off <<= 1)
                tm = fmaxf(tm, __shfl_xor(tm, off, 32));
            float nm = fmaxf(mrow[r], tm);
            alpha[r] = __expf(mrow[r] - nm);
            mrow[r] = nm;
            float p0 = __expf(v0 - nm);
            float p1 = __expf(v1 - nm);
            float ts = p0 + p1;
            #pragma unroll
            for (int off = 1; off < 16; off <<= 1)
                ts += __shfl_xor(ts, off, 32);
            lrow[r] = lrow[r] * alpha[r] + ts;
            int rl = lg ? 8 + r : r;
            Ps[wid][rl][l16]      = (_Float16)p0;
            Ps[wid][rl][16 + l16] = (_Float16)p1;
        }

        // rescale running output
        #pragma unroll
        for (int n = 0; n < 4; ++n)
            #pragma unroll
            for (int r = 0; r < 8; ++r) o[n][r] *= alpha[r];

        // P (16x32) as A fragment from per-wave LDS (same-wave LDS is in-order)
        const _Float16* pr = &Ps[wid][l16][0];
        v16h ap = frag_cat(ld8(pr + lg * 8), ld8(pr + 16 + lg * 8));

        #pragma unroll
        for (int n = 0; n < 4; ++n) {
            const _Float16* vr = &Vt[n * 16 + l16][0];
            v16h bv = frag_cat(ld8(vr + lg * 8), ld8(vr + 16 + lg * 8));
            o[n] = __builtin_amdgcn_wmma_f32_16x16x32_f16(
                false, ap, false, bv, (short)0, o[n], false, false);
        }
    }

    // finalize: divide by row sum, store f16 for the wo GEMM
    #pragma unroll
    for (int n = 0; n < 4; ++n)
        #pragma unroll
        for (int r = 0; r < 8; ++r) {
            int row = q0 + (lg ? 8 + r : r);
            size_t idx = ((size_t)bb * SEQ + row) * DIM + hh * HD + n * 16 + l16;
            O[idx] = (_Float16)(o[n][r] / lrow[r]);
        }
}

// ---------------------------------------------------------------------------
// Elementwise fusions
__global__ __launch_bounds__(256)
void ew_residual_superpos(const float* __restrict__ x, const float* __restrict__ proj,
                          const float* __restrict__ sphase, float* __restrict__ x1,
                          int total) {
    int i = blockIdx.x * 256 + threadIdx.x;
    if (i < total) {
        float t = x[i] + proj[i];
        float ph = sphase[i & (DIM - 1)];
        x1[i] = t * __cosf(ph * t);
    }
}

__global__ __launch_bounds__(256)
void ew_residual2(const float* __restrict__ x1, const float* __restrict__ ff,
                  float* __restrict__ x2, _Float16* __restrict__ x2h, int total) {
    int i = blockIdx.x * 256 + threadIdx.x;
    if (i < total) {
        float t = x1[i] + ff[i];
        x2[i] = t;
        x2h[i] = (_Float16)t;
    }
}

__global__ __launch_bounds__(256)
void ew_gate(const float* __restrict__ p, const float* __restrict__ gl,
             const float* __restrict__ x2, float* __restrict__ out, int total) {
    int i = blockIdx.x * 256 + threadIdx.x;
    if (i < total) {
        float g = 1.f / (1.f + __expf(-gl[i]));
        out[i] = p[i] * g + x2[i] * (1.f - g);
    }
}

// ---------------------------------------------------------------------------
extern "C" void kernel_launch(void* const* d_in, const int* in_sizes, int n_in,
                              void* d_out, int out_size, void* d_ws, size_t ws_size,
                              hipStream_t stream) {
    (void)in_sizes; (void)n_in; (void)out_size; (void)ws_size;
    const float* x      = (const float*)d_in[0];
    const float* wq     = (const float*)d_in[1];
    const float* bq     = (const float*)d_in[2];
    const float* wk     = (const float*)d_in[3];
    const float* bk     = (const float*)d_in[4];
    const float* wv     = (const float*)d_in[5];
    const float* bv     = (const float*)d_in[6];
    const float* wo     = (const float*)d_in[7];
    const float* bo     = (const float*)d_in[8];
    const float* qphase = (const float*)d_in[9];
    const float* sphase = (const float*)d_in[10];
    const float* wi_p   = (const float*)d_in[11];
    const float* bi_p   = (const float*)d_in[12];
    const float* wi_g   = (const float*)d_in[13];
    const float* bi_g   = (const float*)d_in[14];
    const float* ln1_g  = (const float*)d_in[15];
    const float* ln1_b  = (const float*)d_in[16];
    const float* ln2_g  = (const float*)d_in[17];
    const float* ln2_b  = (const float*)d_in[18];
    const float* w1     = (const float*)d_in[19];
    const float* b1     = (const float*)d_in[20];
    const float* w2     = (const float*)d_in[21];
    const float* b2     = (const float*)d_in[22];
    float* out = (float*)d_out;

    // workspace layout (bytes), total 112 MB
    char* ws = (char*)d_ws;
    const size_t MB = 1024ull * 1024ull;
    _Float16* h16    = (_Float16*)(ws);              // 8 MB (ln1 out, reused ln2)
    _Float16* q16    = (_Float16*)(ws + 8  * MB);    // 8 MB (part of 32MB run)
    _Float16* k16    = (_Float16*)(ws + 16 * MB);    // 8 MB (part of 32MB run)
    _Float16* v16    = (_Float16*)(ws + 24 * MB);    // 8 MB (part of 32MB run)
    _Float16* attn16 = (_Float16*)(ws + 32 * MB);    // 8 MB (part of 32MB run)
    _Float16* ff16   = (_Float16*)(ws + 8  * MB);    // 32 MB alias over q/k/v/attn
    float*    proj   = (float*)(ws + 40 * MB);       // 16 MB (also ffout, gbuf)
    float*    x1     = (float*)(ws + 56 * MB);       // 16 MB
    float*    x2     = (float*)(ws + 72 * MB);       // 16 MB
    _Float16* x2h    = (_Float16*)(ws + 88 * MB);    // 8 MB
    float*    pbuf   = (float*)(ws + 96 * MB);       // 16 MB

    const int total = MROWS * DIM;
    const dim3 blk(256);
    const dim3 gemm_g1(DIM / 64, MROWS / 128);     // N=1024
    const dim3 gemm_gF(FFDIM / 64, MROWS / 128);   // N=4096
    const dim3 ew_g((total + 255) / 256);

    // 1. LN1
    layernorm_f16_kernel<<<MROWS, blk, 0, stream>>>(x, ln1_g, ln1_b, h16);
    // 2. QKV projections (f16 outputs for attention)
    gemm_f16_kernel<0><<<gemm_g1, blk, 0, stream>>>(h16, wq, bq, nullptr, q16, DIM, DIM);
    gemm_f16_kernel<0><<<gemm_g1, blk, 0, stream>>>(h16, wk, bk, nullptr, k16, DIM, DIM);
    gemm_f16_kernel<0><<<gemm_g1, blk, 0, stream>>>(h16, wv, bv, nullptr, v16, DIM, DIM);
    // 3. flash attention with cosine modulation
    flash_attn_kernel<<<dim3(SEQ / 128, NHEADS, BATCH), blk, 0, stream>>>(
        q16, k16, v16, qphase, attn16);
    // 4. output projection
    gemm_f16_kernel<0><<<gemm_g1, blk, 0, stream>>>(attn16, wo, bo, proj, nullptr, DIM, DIM);
    // 5. residual + superposition
    ew_residual_superpos<<<ew_g, blk, 0, stream>>>(x, proj, sphase, x1, total);
    // 6. LN2
    layernorm_f16_kernel<<<MROWS, blk, 0, stream>>>(x1, ln2_g, ln2_b, h16);
    // 7. FFN up + GELU (f16 out)
    gemm_f16_kernel<1><<<gemm_gF, blk, 0, stream>>>(h16, w1, b1, nullptr, ff16, DIM, FFDIM);
    // 8. FFN down (f32 out into proj)
    gemm_f16_kernel<0><<<gemm_g1, blk, 0, stream>>>(ff16, w2, b2, proj, nullptr, FFDIM, DIM);
    // 9. residual 2 (+f16 copy for gate GEMMs)
    ew_residual2<<<ew_g, blk, 0, stream>>>(x1, proj, x2, x2h, total);
    // 10. gate projections
    gemm_f16_kernel<0><<<gemm_g1, blk, 0, stream>>>(x2h, wi_p, bi_p, pbuf, nullptr, DIM, DIM);
    gemm_f16_kernel<0><<<gemm_g1, blk, 0, stream>>>(x2h, wi_g, bi_g, proj, nullptr, DIM, DIM);
    // 11. final gated mix
    ew_gate<<<ew_g, blk, 0, stream>>>(pbuf, proj, x2, out, total);
}
